// SelfAttentiveBimodalFusion_51393578664164
// MI455X (gfx1250) — matrix operand order repaired and checked by reading, not compiled, not run on hardware
//
#include <hip/hip_runtime.h>
#include <hip/hip_bf16.h>

// ---------------------------------------------------------------------------
// Problem constants (from reference)
// ---------------------------------------------------------------------------
#define BATCH   2
#define NPER    8192            // points per sample
#define NPTS    (BATCH * NPER)  // 16384
#define KNB     16              // neighbors
#define CIN     128             // in_main == in_mod == emb
#define CC      256             // C = 2*EMB
#define CS      32              // C / share_planes

// ---------------------------------------------------------------------------
// WMMA types / helpers (CDNA5, wave32, v_wmma_f32_16x16x32_bf16)
// ---------------------------------------------------------------------------
typedef __attribute__((ext_vector_type(16))) __bf16        v16bf;
typedef __attribute__((ext_vector_type(8)))  float         v8f;
typedef __attribute__((ext_vector_type(4)))  unsigned int  v4u;
typedef __attribute__((ext_vector_type(4)))  int           v4i;

union BfFrag {
    unsigned int u[8];
    v4u          q[2];
    v16bf        v;
};

__device__ __forceinline__ unsigned short f2bf(float f) {
    unsigned int u = __builtin_bit_cast(unsigned int, f);
    u += 0x7FFFu + ((u >> 16) & 1u);       // round-to-nearest-even
    return (unsigned short)(u >> 16);
}

// ---------------------------------------------------------------------------
// Async global->LDS staging (gfx1250 GLOBAL_LOAD_ASYNC_TO_LDS_B128 + ASYNCcnt)
// Builtin parameter types are v4i addrspace(1)* / v4i addrspace(3)* (from the
// round-2 diagnostic); LDS generic pointers carry the LDS offset in the low
// 32 bits, addrspace(3) pointers are 32-bit.
// ---------------------------------------------------------------------------
#if __has_builtin(__builtin_amdgcn_global_load_async_to_lds_b128)
#define HAVE_ASYNC_LDS 1
typedef __attribute__((address_space(1))) v4i* g_v4i;
typedef __attribute__((address_space(3))) v4i* l_v4i;
#endif

__device__ __forceinline__ void async_g2l_b128(const float* g, float* l) {
#ifdef HAVE_ASYNC_LDS
    __builtin_amdgcn_global_load_async_to_lds_b128(
        (g_v4i)(unsigned long long)(uintptr_t)g,
        (l_v4i)(unsigned int)(uintptr_t)l, 0, 0);
#else
    *(v4u*)l = *(const v4u*)g;
#endif
}

__device__ __forceinline__ void wait_async0() {
#ifdef HAVE_ASYNC_LDS
#if __has_builtin(__builtin_amdgcn_s_wait_asynccnt)
    __builtin_amdgcn_s_wait_asynccnt(0);
#else
    asm volatile("s_wait_asynccnt 0x0" ::: "memory");
#endif
#endif
}

// ---------------------------------------------------------------------------
// fp32 -> bf16 conversion pass (activations only; weights get prepacked)
// ---------------------------------------------------------------------------
__global__ void cvt_bf16_kernel(const float* __restrict__ src,
                                unsigned short* __restrict__ dst, int n) {
    int i = blockIdx.x * blockDim.x + threadIdx.x;
    int stride = gridDim.x * blockDim.x;
    for (; i < n; i += stride) dst[i] = f2bf(src[i]);
}

// ---------------------------------------------------------------------------
// Prepack a (Kdim x N) f32 weight matrix into WMMA B-fragment order:
//   out[(((nt*kchunks + kc)*32 + lane)*8 + v] = pack(bf16 W[k,n], bf16 W[k+1,n])
// with n = nt*16 + (lane&15), h = lane>>4, k = kc*32 + v*2 + h*8 + (v>>2)*8.
// Each wave then reads its fragment as two contiguous b128 loads.
// ---------------------------------------------------------------------------
__global__ void prepack_b_kernel(const float* __restrict__ W, int N, int Kdim,
                                 unsigned int* __restrict__ out) {
    const int kchunks = Kdim >> 5;
    const int total = (N >> 4) * kchunks * 256;
    int i = blockIdx.x * blockDim.x + threadIdx.x;
    int stride = gridDim.x * blockDim.x;
    for (; i < total; i += stride) {
        int v    = i & 7;
        int lane = (i >> 3) & 31;
        int kc   = (i >> 8) % kchunks;
        int nt   = i / (kchunks << 8);
        int h = lane >> 4;
        int n = (nt << 4) + (lane & 15);
        int k = (kc << 5) + (v << 1) + (h << 3) + ((v >> 2) << 3);
        unsigned int lo = f2bf(W[(size_t)k * N + n]);
        unsigned int hi = f2bf(W[(size_t)(k + 1) * N + n]);
        out[i] = lo | (hi << 16);
    }
}

// ---------------------------------------------------------------------------
// bf16 WMMA GEMM with prepacked B: C[m, col_base+n] = A@B + bias[n]
// 128 threads = 4 waves, each wave owns one 16x16 tile (4 M-tiles per block).
// ---------------------------------------------------------------------------
__global__ void wmma_gemm_bias_kernel(const unsigned short* __restrict__ A, int lda,
                                      const unsigned int* __restrict__ Bp,
                                      const float* __restrict__ bias, int Kdim,
                                      float* __restrict__ Cf,
                                      unsigned short* __restrict__ Cb,
                                      int ldc, int col_base) {
    const int tid  = threadIdx.x;
    const int lane = tid & 31;
    const int ml   = lane & 15;
    const int h    = lane >> 4;
    const int m0   = (blockIdx.x * (blockDim.x >> 5) + (tid >> 5)) << 4;
    const int nt   = blockIdx.y;
    const int kchunks = Kdim >> 5;

    v8f acc = (v8f)0.0f;
    for (int kc = 0; kc < kchunks; ++kc) {
        BfFrag a, b;
        const unsigned short* arow =
            A + (size_t)(m0 + ml) * lda + (kc << 5) + (h << 3);
        a.q[0] = *(const v4u*)(arow);
        a.q[1] = *(const v4u*)(arow + 16);
        const unsigned int* bp = Bp + ((size_t)(nt * kchunks + kc) * 32 + lane) * 8;
        b.q[0] = *(const v4u*)(bp);
        b.q[1] = *(const v4u*)(bp + 4);
        acc = __builtin_amdgcn_wmma_f32_16x16x32_bf16(
                false, a.v, false, b.v, (short)0, acc, false, false);
    }

    const int col = col_base + (nt << 4) + ml;
    const float bs = bias[(nt << 4) + ml];
#pragma unroll
    for (int r = 0; r < 8; ++r) {
        int m = m0 + r + (h << 3);
        float val = acc[r] + bs;
        Cf[(size_t)m * ldc + col] = val;
        if (Cb) Cb[(size_t)m * ldc + col] = f2bf(val);
    }
}

// ---------------------------------------------------------------------------
// Brute-force per-sample kNN (top-16 by squared distance, self included)
// 256 queries per block; candidate tiles staged through LDS; next tile is
// prefetched (global_prefetch_b8).
// ---------------------------------------------------------------------------
__global__ void knn_kernel(const float* __restrict__ xyz, int* __restrict__ idx) {
    __shared__ float sx[256], sy[256], sz[256];
    const int tid = threadIdx.x;
    const int q   = blockIdx.x * 256 + tid;       // 8192 % 256 == 0 -> block in one sample
    const int b   = q >> 13;
    const int base = b << 13;

    const float px = xyz[q * 3 + 0];
    const float py = xyz[q * 3 + 1];
    const float pz = xyz[q * 3 + 2];

    float bd[KNB];
    int   bi[KNB];
#pragma unroll
    for (int i = 0; i < KNB; ++i) { bd[i] = 3.4e38f; bi[i] = base; }

    for (int t = 0; t < NPER / 256; ++t) {
        int c = base + t * 256 + tid;
        sx[tid] = xyz[c * 3 + 0];
        sy[tid] = xyz[c * 3 + 1];
        sz[tid] = xyz[c * 3 + 2];
        if (t + 1 < NPER / 256)
            __builtin_prefetch(&xyz[(c + 256) * 3], 0, 3);
        __syncthreads();
        for (int j = 0; j < 256; ++j) {
            float dx = px - sx[j], dy = py - sy[j], dz = pz - sz[j];
            float d = dx * dx + dy * dy + dz * dz;
            if (d < bd[KNB - 1]) {
                int ci = base + t * 256 + j;
                int pos = KNB - 1;
#pragma unroll
                for (int s = KNB - 2; s >= 0; --s) {
                    if (d < bd[s]) { bd[s + 1] = bd[s]; bi[s + 1] = bi[s]; pos = s; }
                }
                bd[pos] = d; bi[pos] = ci;
            }
        }
        __syncthreads();
    }
#pragma unroll
    for (int i = 0; i < KNB; ++i) idx[q * KNB + i] = bi[i];
}

// ---------------------------------------------------------------------------
// Fused per-point attention. One block (64 threads = 2 waves) per point.
//  stage 0: async-stage gathered k/v rows (16x256 f32 each) into LDS
//  stage 1: logits w (16x256 bf16) and vp = v+p_r (in place over staged v)
//  stage 2: WMMA 16x256 @ 256x32 (prepacked Ww1) + bias + BN + ReLU -> bf16
//  stage 3: WMMA 16x32  @ 32x32  (prepacked Ww2) + bias             -> logits
//  stage 4: softmax over the 16 neighbors per CS column
//  stage 5: agg[c] = sum_j vp[j,c]*sm[j,c&31] ; out = agg + x (residual)
// ---------------------------------------------------------------------------
__global__ void attn_kernel(const float* __restrict__ qf,
                            const float* __restrict__ kf,
                            const float* __restrict__ vf,
                            const int*   __restrict__ idx,
                            const float* __restrict__ xyz,
                            const float* __restrict__ xres,
                            const float* __restrict__ Wp1, const float* __restrict__ bp1,
                            const float* __restrict__ gp,  const float* __restrict__ betap,
                            const float* __restrict__ Wp2, const float* __restrict__ bp2,
                            const float* __restrict__ gw1, const float* __restrict__ betaw1,
                            const unsigned int* __restrict__ Bp1,   // prepacked Ww1
                            const float* __restrict__ bw1,
                            const float* __restrict__ gw2, const float* __restrict__ betaw2,
                            const unsigned int* __restrict__ Bp2,   // prepacked Ww2
                            const float* __restrict__ bw2,
                            float* __restrict__ out) {
    __shared__ float          s_wp2[3 * CC];
    __shared__ float          s_bp2[CC], s_gw1[CC], s_bt1[CC], s_q[CC];
    __shared__ float          s_pr3[KNB * 3];
    __shared__ int            s_idx[KNB];
    __shared__ float          s_kg[KNB * CC];     // staged gathered k rows
    __shared__ float          s_vg[KNB * CC];     // staged gathered v rows -> vp in place
    __shared__ unsigned short s_w[KNB * CC];      // bf16 logits (A of GEMM1)
    __shared__ unsigned short s_a2[KNB * CS];     // bf16 A of GEMM2
    __shared__ float          s_logit[KNB * CS];
    __shared__ float          s_sm[KNB * CS];

    const int tid = threadIdx.x;
    const int n   = blockIdx.x;

    for (int c = tid; c < CC; c += 64) {
        s_wp2[c]          = Wp2[c];
        s_wp2[CC + c]     = Wp2[CC + c];
        s_wp2[2 * CC + c] = Wp2[2 * CC + c];
        s_bp2[c] = bp2[c];
        s_gw1[c] = gw1[c];
        s_bt1[c] = betaw1[c];
        s_q[c]   = qf[(size_t)n * CC + c];
    }
    if (tid < KNB) s_idx[tid] = idx[n * KNB + tid];
    __syncthreads();

    // stage 0: async copy the 16 gathered k/v rows into LDS (b128 granules)
    for (int t = tid; t < KNB * (CC / 4); t += 64) {
        int j = t >> 6, p = (t & 63) << 2;       // row j, float offset p (16B aligned)
        int ix = s_idx[j];
        async_g2l_b128(kf + (size_t)ix * CC + p, &s_kg[j * CC + p]);
        async_g2l_b128(vf + (size_t)ix * CC + p, &s_vg[j * CC + p]);
    }

    // relative-position mini-MLP first stage: 3 -> 3, BN, ReLU (overlaps copies)
    if (tid < KNB) {
        int j = tid, ix = s_idx[j];
        float p0 = xyz[ix * 3 + 0] - xyz[n * 3 + 0];
        float p1 = xyz[ix * 3 + 1] - xyz[n * 3 + 1];
        float p2 = xyz[ix * 3 + 2] - xyz[n * 3 + 2];
#pragma unroll
        for (int o = 0; o < 3; ++o) {
            float t = p0 * Wp1[o] + p1 * Wp1[3 + o] + p2 * Wp1[6 + o] + bp1[o];
            t = gp[o] * t + betap[o];
            s_pr3[j * 3 + o] = t > 0.0f ? t : 0.0f;
        }
    }
    wait_async0();
    __syncthreads();

    // stage 1: logits + (v + p_r) in place
    for (int t = tid; t < KNB * CC; t += 64) {
        int j = t >> 8, c = t & (CC - 1);
        float pr = s_pr3[j * 3 + 0] * s_wp2[c]
                 + s_pr3[j * 3 + 1] * s_wp2[CC + c]
                 + s_pr3[j * 3 + 2] * s_wp2[2 * CC + c] + s_bp2[c];
        float wv = s_kg[t] - s_q[c] + pr;
        float wl = s_gw1[c] * wv + s_bt1[c];
        wl = wl > 0.0f ? wl : 0.0f;
        s_w[t]  = f2bf(wl);
        s_vg[t] = s_vg[t] + pr;
    }
    __syncthreads();

    // stage 2: GEMM1 (16x256)@(256x32); wave w owns output cols [16w,16w+16)
    const int lane = tid & 31;
    const int ml   = lane & 15;
    const int h    = lane >> 4;
    const int nt   = tid >> 5;
    const int col  = (nt << 4) + ml;

    v8f acc = (v8f)0.0f;
    for (int kc = 0; kc < CC / 32; ++kc) {
        BfFrag a, b;
        const unsigned short* arow = s_w + ml * CC + (kc << 5) + (h << 3);
        a.q[0] = *(const v4u*)(arow);
        a.q[1] = *(const v4u*)(arow + 16);
        const unsigned int* bp = Bp1 + ((size_t)(nt * 8 + kc) * 32 + lane) * 8;
        b.q[0] = *(const v4u*)(bp);
        b.q[1] = *(const v4u*)(bp + 4);
        acc = __builtin_amdgcn_wmma_f32_16x16x32_bf16(
                false, a.v, false, b.v, (short)0, acc, false, false);
    }
    {
        float bb = bw1[col], g2 = gw2[col], b2 = betaw2[col];
#pragma unroll
        for (int r = 0; r < 8; ++r) {
            float t = g2 * (acc[r] + bb) + b2;
            t = t > 0.0f ? t : 0.0f;
            s_a2[(r + (h << 3)) * CS + col] = f2bf(t);
        }
    }
    __syncthreads();

    // stage 3: GEMM2 (16x32)@(32x32)
    v8f acc2 = (v8f)0.0f;
    {
        BfFrag a, b;
        const unsigned short* arow = s_a2 + ml * CS + (h << 3);
        a.q[0] = *(const v4u*)(arow);
        a.q[1] = *(const v4u*)(arow + 16);
        const unsigned int* bp = Bp2 + ((size_t)nt * 32 + lane) * 8;
        b.q[0] = *(const v4u*)(bp);
        b.q[1] = *(const v4u*)(bp + 4);
        acc2 = __builtin_amdgcn_wmma_f32_16x16x32_bf16(
                false, a.v, false, b.v, (short)0, acc2, false, false);
    }
    {
        float bb = bw2[col];
#pragma unroll
        for (int r = 0; r < 8; ++r)
            s_logit[(r + (h << 3)) * CS + col] = acc2[r] + bb;
    }
    __syncthreads();

    // stage 4: softmax over the 16 neighbors, per CS column
    if (tid < CS) {
        float mx = -3.4e38f;
#pragma unroll
        for (int m = 0; m < KNB; ++m) mx = fmaxf(mx, s_logit[m * CS + tid]);
        float sum = 0.0f;
#pragma unroll
        for (int m = 0; m < KNB; ++m) {
            float e = __expf(s_logit[m * CS + tid] - mx);
            s_sm[m * CS + tid] = e;
            sum += e;
        }
        float inv = 1.0f / sum;
#pragma unroll
        for (int m = 0; m < KNB; ++m) s_sm[m * CS + tid] *= inv;
    }
    __syncthreads();

    // stage 5: weighted aggregation + residual
    for (int c = tid; c < CC; c += 64) {
        int cs = c & (CS - 1);
        float a = 0.0f;
#pragma unroll
        for (int j = 0; j < KNB; ++j)
            a += s_vg[j * CC + c] * s_sm[j * CS + cs];
        out[(size_t)n * CC + c] = a + xres[(size_t)n * CC + c];
    }
}

// ---------------------------------------------------------------------------
// Host launcher
// ---------------------------------------------------------------------------
extern "C" void kernel_launch(void* const* d_in, const int* in_sizes, int n_in,
                              void* d_out, int out_size, void* d_ws, size_t ws_size,
                              hipStream_t stream) {
    const float* x_main = (const float*)d_in[0];
    const float* x_mod  = (const float*)d_in[1];
    const float* xyz    = (const float*)d_in[2];
    const float* We2d   = (const float*)d_in[3];
    const float* be2d   = (const float*)d_in[4];
    const float* We3d   = (const float*)d_in[5];
    const float* be3d   = (const float*)d_in[6];
    const float* Wq     = (const float*)d_in[7];
    const float* bq     = (const float*)d_in[8];
    const float* Wk     = (const float*)d_in[9];
    const float* bk     = (const float*)d_in[10];
    const float* Wv     = (const float*)d_in[11];
    const float* bv     = (const float*)d_in[12];
    const float* Wp1    = (const float*)d_in[13];
    const float* bp1    = (const float*)d_in[14];
    const float* gp     = (const float*)d_in[15];
    const float* betap  = (const float*)d_in[16];
    const float* Wp2    = (const float*)d_in[17];
    const float* bp2    = (const float*)d_in[18];
    const float* gw1    = (const float*)d_in[19];
    const float* betaw1 = (const float*)d_in[20];
    const float* Ww1    = (const float*)d_in[21];
    const float* bw1    = (const float*)d_in[22];
    const float* gw2    = (const float*)d_in[23];
    const float* betaw2 = (const float*)d_in[24];
    const float* Ww2    = (const float*)d_in[25];
    const float* bw2    = (const float*)d_in[26];
    float* out = (float*)d_out;

    // -------- workspace carve-up --------
    char* w = (char*)d_ws;
    size_t off = 0;
    auto alloc = [&](size_t bytes) -> char* {
        char* p = w + off;
        off += (bytes + 255) & ~(size_t)255;
        return p;
    };
    unsigned short* xb_main = (unsigned short*)alloc((size_t)NPTS * CIN * 2);
    unsigned short* xb_mod  = (unsigned short*)alloc((size_t)NPTS * CIN * 2);
    float*          x_f32   = (float*)alloc((size_t)NPTS * CC * 4);
    unsigned short* x_bf16  = (unsigned short*)alloc((size_t)NPTS * CC * 2);
    float*          q_f32   = (float*)alloc((size_t)NPTS * CC * 4);
    float*          k_f32   = (float*)alloc((size_t)NPTS * CC * 4);
    float*          v_f32   = (float*)alloc((size_t)NPTS * CC * 4);
    int*            idx     = (int*)alloc((size_t)NPTS * KNB * 4);
    // prepacked B fragments (dwords): (N/16) * (K/32) * 256
    unsigned int* bp_e3d = (unsigned int*)alloc((size_t)(CIN/16)*(CIN/32)*256*4);
    unsigned int* bp_e2d = (unsigned int*)alloc((size_t)(CIN/16)*(CIN/32)*256*4);
    unsigned int* bp_q   = (unsigned int*)alloc((size_t)(CC/16)*(CC/32)*256*4);
    unsigned int* bp_k   = (unsigned int*)alloc((size_t)(CC/16)*(CC/32)*256*4);
    unsigned int* bp_v   = (unsigned int*)alloc((size_t)(CC/16)*(CC/32)*256*4);
    unsigned int* bp_w1  = (unsigned int*)alloc((size_t)(CS/16)*(CC/32)*256*4);
    unsigned int* bp_w2  = (unsigned int*)alloc((size_t)(CS/16)*(CS/32)*256*4);
    (void)ws_size; (void)in_sizes; (void)n_in; (void)out_size;

    // -------- 0a. activation bf16 conversions --------
    auto cvt = [&](const float* s, unsigned short* d, int n) {
        int blocks = (n + 255) / 256;
        if (blocks > 1024) blocks = 1024;
        cvt_bf16_kernel<<<blocks, 256, 0, stream>>>(s, d, n);
    };
    cvt(x_main, xb_main, NPTS * CIN);
    cvt(x_mod,  xb_mod,  NPTS * CIN);

    // -------- 0b. weight prepack into B-fragment order --------
    auto prep = [&](const float* W, int N, int K, unsigned int* o) {
        int total = (N / 16) * (K / 32) * 256;
        int blocks = (total + 255) / 256;
        prepack_b_kernel<<<blocks, 256, 0, stream>>>(W, N, K, o);
    };
    prep(We3d, CIN, CIN, bp_e3d);
    prep(We2d, CIN, CIN, bp_e2d);
    prep(Wq,   CC,  CC,  bp_q);
    prep(Wk,   CC,  CC,  bp_k);
    prep(Wv,   CC,  CC,  bp_v);
    prep(Ww1,  CS,  CC,  bp_w1);
    prep(Ww2,  CS,  CS,  bp_w2);

    // -------- 1. embeddings: x = [x_main@We3d+b | x_mod@We2d+b] --------
    {
        dim3 grid(NPTS / 64, CIN / 16);
        wmma_gemm_bias_kernel<<<grid, 128, 0, stream>>>(
            xb_main, CIN, bp_e3d, be3d, CIN, x_f32, x_bf16, CC, 0);
        wmma_gemm_bias_kernel<<<grid, 128, 0, stream>>>(
            xb_mod, CIN, bp_e2d, be2d, CIN, x_f32, x_bf16, CC, CIN);
    }

    // -------- 2. q, k, v = x @ W + b --------
    {
        dim3 grid(NPTS / 64, CC / 16);
        wmma_gemm_bias_kernel<<<grid, 128, 0, stream>>>(
            x_bf16, CC, bp_q, bq, CC, q_f32, nullptr, CC, 0);
        wmma_gemm_bias_kernel<<<grid, 128, 0, stream>>>(
            x_bf16, CC, bp_k, bk, CC, k_f32, nullptr, CC, 0);
        wmma_gemm_bias_kernel<<<grid, 128, 0, stream>>>(
            x_bf16, CC, bp_v, bv, CC, v_f32, nullptr, CC, 0);
    }

    // -------- 3. per-sample kNN --------
    knn_kernel<<<NPTS / 256, 256, 0, stream>>>(xyz, idx);

    // -------- 4. fused attention + residual --------
    attn_kernel<<<NPTS, 64, 0, stream>>>(
        q_f32, k_f32, v_f32, idx, xyz, x_f32,
        Wp1, bp1, gp, betap, Wp2, bp2, gw1, betaw1,
        bp_w1, bw1, gw2, betaw2, bp_w2, bw2, out);
}
